// SaliencyPruner_72722386256515
// MI455X (gfx1250) — matrix operand order repaired and checked by reading, not compiled, not run on hardware
//
#include <hip/hip_runtime.h>
#include <hip/hip_bf16.h>
#include <math.h>

// ---------------------------------------------------------------------------
// Problem constants: B=4, N=16384, D=1024, H=256, k = int(16384*0.7) = 11468
// ---------------------------------------------------------------------------
#define BB 4
#define NN 16384
#define DD 1024
#define HH 256
#define KK 11468
#define ROWS (BB * NN)   // 65536 flattened rows

typedef __attribute__((ext_vector_type(16))) __bf16    v16bf;
typedef __attribute__((ext_vector_type(8)))  __bf16    v8bf;
typedef __attribute__((ext_vector_type(8)))  float     v8f;
typedef __attribute__((ext_vector_type(4)))  unsigned  u32x4;
typedef __attribute__((ext_vector_type(8)))  int       i32x8;
typedef __attribute__((ext_vector_type(4)))  int       i32x4;

__device__ __forceinline__ __bf16 tobf(float f) {
  unsigned u = __float_as_uint(f);
  unsigned r = u + 0x7FFFu + ((u >> 16) & 1u);   // round-to-nearest-even
  unsigned short h = (unsigned short)(r >> 16);
  return __builtin_bit_cast(__bf16, h);
}

// ---------------------------------------------------------------------------
// Kernel 0: W1 [D=1024, H=256] f32 -> W1bf [H=256][D=1024] bf16 (transposed)
// so the TDM can stage K-slabs as raw bytes (no conversion in the hot loop).
// ---------------------------------------------------------------------------
__global__ __launch_bounds__(256) void convert_w1_kernel(
    const float* __restrict__ W1, __bf16* __restrict__ W1bf) {
  int t = blockIdx.x * 256 + threadIdx.x;   // grid 1024 -> 262144 elements
  int c = t & 255;
  int k = t >> 8;
  W1bf[c * DD + k] = tobf(W1[k * HH + c]);
}

// ---------------------------------------------------------------------------
// TDM stage: copy a 256-row x 32-col bf16 tile (row stride DD elements) from
// global into LDS, inserting 4 pad DWORDs after every 16 DWORDs (one 32-elem
// row = 64B), i.e. LDS row stride = 40 bf16 (bank-spread for B-frag reads).
// ---------------------------------------------------------------------------
#define SLAB_STRIDE 40            // bf16 elements per staged row in LDS
#define SLAB_ELEMS (256 * SLAB_STRIDE)

__device__ __forceinline__ void tdm_stage(__bf16* lds_dst, const __bf16* gsrc) {
  unsigned lds_addr = (unsigned)(uintptr_t)lds_dst;   // low 32 bits = LDS offset
  unsigned long long ga = (unsigned long long)(uintptr_t)gsrc;
  // D# group 0: count=1 | lds_addr | global_addr[56:0] | type=2
  u32x4 g0 = { 1u,
               lds_addr,
               (unsigned)ga,
               (unsigned)((ga >> 32) & 0x01FFFFFFu) | (2u << 30) };
  // D# group 1: data_size=1(2B), pad_enable, pad_interval=3(16 dw),
  //             pad_amount=3(4 dw); tensor_dim0=1024, tensor_dim1=256,
  //             tile_dim0=32, tile_dim1=256, tensor_dim0_stride=1024
  i32x8 g1 = { (int)((1u << 16) | (1u << 20) | (3u << 22) | (3u << 25)),
               (int)(1024u << 16),   // tensor_dim0[15:0] -> bits [63:48]
               (int)(256u << 16),    // tensor_dim0 hi=0 | tensor_dim1[15:0]
               (int)(32u << 16),     // tensor_dim1 hi=0 | tile_dim0=32
               (int)256,             // tile_dim1=256 | tile_dim2=0
               (int)1024,            // tensor_dim0_stride[31:0]
               0, 0 };
  i32x4 z4 = { 0, 0, 0, 0 };
#if __has_include(<hip/amd_detail/amd_gfx1250_TDM.h>)
  i32x8 z8 = { 0, 0, 0, 0, 0, 0, 0, 0 };
  __builtin_amdgcn_tensor_load_to_lds(g0, g1, z4, z4, z8, 0);
#else
  __builtin_amdgcn_tensor_load_to_lds(g0, g1, z4, z4, 0);
#endif
}

// ---------------------------------------------------------------------------
// Kernel 1: scores = sigmoid(relu(x@W1 + b1) @ W2 + b2)
// 8 waves/block, each wave owns a 16-row strip and ALL 16 column tiles
// (16 f32 accumulator tiles). K-outer loop: W1 bf16 K-slab double-buffered
// into LDS by the Tensor Data Mover, overlapped with 16 WMMAs per chunk.
// B fragments are software-pipelined one tile ahead so LDS loads overlap
// the matrix pipe instead of serializing on s_wait_dscnt 0.
// ---------------------------------------------------------------------------
__device__ __forceinline__ v16bf load_bfrag(const __bf16* sb, int ct, int col,
                                            int half) {
  const __bf16* bp = sb + (ct * 16 + col) * SLAB_STRIDE + half * 8;
  v8bf lo = *(const v8bf*)(bp);
  v8bf hi = *(const v8bf*)(bp + 16);
  v16bf bf;
#pragma unroll
  for (int j = 0; j < 8; ++j) { bf[j] = lo[j]; bf[8 + j] = hi[j]; }
  return bf;
}

__global__ __launch_bounds__(256) void scores_kernel(
    const float* __restrict__ x, const __bf16* __restrict__ W1bf,
    const float* __restrict__ b1, const float* __restrict__ W2,
    const float* __restrict__ b2, float* __restrict__ scores) {
  __shared__ __bf16 slab[2][SLAB_ELEMS];   // 2 x 20 KB

  const int tid  = threadIdx.x;
  const int lane = tid & 31;
  const int wave = tid >> 5;
  const int col  = lane & 15;   // A-row (M) and B/C/D lane column (N)
  const int half = lane >> 4;
  const int rowbase = blockIdx.x * 128 + wave * 16;
  const float* xrow = x + (size_t)(rowbase + col) * DD;

  v8f acc[16];
#pragma unroll
  for (int ct = 0; ct < 16; ++ct) acc[ct] = {};

  // Prologue: DMA first K-slab into buffer 0 (one TDM issue for the block).
  if (tid == 0) tdm_stage(&slab[0][0], W1bf);

  for (int ks = 0; ks < 32; ++ks) {
    const int k0 = ks * 32;
    if (tid == 0) {
      if (ks < 31) {
        tdm_stage(&slab[(ks + 1) & 1][0], W1bf + (k0 + 32));
        __builtin_amdgcn_s_wait_tensorcnt(1);   // current slab complete
      } else {
        __builtin_amdgcn_s_wait_tensorcnt(0);
      }
    }
    __syncthreads();
    const __bf16* sb = &slab[ks & 1][0];

    // ---- A fragment: 16x32 bf16 (lane: M=col, K per ISA half/j layout)
    const float* ap = xrow + k0 + half * 8;
    float4 f0 = *(const float4*)(ap);
    float4 f1 = *(const float4*)(ap + 4);
    float4 f2 = *(const float4*)(ap + 16);
    float4 f3 = *(const float4*)(ap + 20);
    if (ks < 31) __builtin_prefetch(ap + 32, 0, 0);   // global_prefetch_b8
    v16bf af;
    af[0]  = tobf(f0.x); af[1]  = tobf(f0.y); af[2]  = tobf(f0.z); af[3]  = tobf(f0.w);
    af[4]  = tobf(f1.x); af[5]  = tobf(f1.y); af[6]  = tobf(f1.z); af[7]  = tobf(f1.w);
    af[8]  = tobf(f2.x); af[9]  = tobf(f2.y); af[10] = tobf(f2.z); af[11] = tobf(f2.w);
    af[12] = tobf(f3.x); af[13] = tobf(f3.y); af[14] = tobf(f3.z); af[15] = tobf(f3.w);

    // ---- 16 column tiles, one WMMA each; B frag pipelined one tile ahead
    v16bf bf_cur = load_bfrag(sb, 0, col, half);
#pragma unroll
    for (int ct = 0; ct < 16; ++ct) {
      v16bf bf_next;
      if (ct < 15) bf_next = load_bfrag(sb, ct + 1, col, half);
      acc[ct] = __builtin_amdgcn_wmma_f32_16x16x32_bf16(
          false, af, false, bf_cur, (short)0, acc[ct], false, false);
      bf_cur = bf_next;
    }
    __syncthreads();   // all waves done with this buffer before TDM reuses it
  }

  // Epilogue: bias + relu + dot(W2), reduce over 16 columns, sigmoid.
  float sacc[8];
#pragma unroll
  for (int v = 0; v < 8; ++v) sacc[v] = 0.0f;
#pragma unroll
  for (int ct = 0; ct < 16; ++ct) {
    const float b1v = b1[ct * 16 + col];
    const float w2v = W2[ct * 16 + col];
#pragma unroll
    for (int v = 0; v < 8; ++v) {
      float h = acc[ct][v] + b1v;
      sacc[v] += (h > 0.0f) ? h * w2v : 0.0f;
    }
  }
#pragma unroll
  for (int v = 0; v < 8; ++v) {
    float s = sacc[v];
    s += __shfl_xor(s, 1);
    s += __shfl_xor(s, 2);
    s += __shfl_xor(s, 4);
    s += __shfl_xor(s, 8);
    sacc[v] = s;
  }
  if (col == 0) {
    const float bb = b2[0];
#pragma unroll
    for (int v = 0; v < 8; ++v) {
      const float z = sacc[v] + bb;
      scores[rowbase + half * 8 + v] = 1.0f / (1.0f + __expf(-z));
    }
  }
}

// ---------------------------------------------------------------------------
// Kernel 2: per-batch top-k (radix-select threshold + ordered compaction),
// ascending kept indices; JAX tie semantics (lowest index wins).
// ---------------------------------------------------------------------------
__device__ __forceinline__ unsigned fkey(float f) {
  unsigned u = __float_as_uint(f);
  return (u & 0x80000000u) ? ~u : (u | 0x80000000u);  // larger float -> larger key
}

__global__ __launch_bounds__(1024) void topk_kernel(
    const float* __restrict__ scores, int* __restrict__ idx) {
  const int b = blockIdx.x;
  const int tid = threadIdx.x;
  const float* sc = scores + b * NN;

  __shared__ unsigned hist[256];
  __shared__ unsigned sh_prefix;
  __shared__ int sh_rem;
  __shared__ int sbuf[1024];
  __shared__ int s_base, s_tie, s_tot_e, s_tot_k;

  if (tid == 0) { sh_prefix = 0u; sh_rem = KK; }
  __syncthreads();

  for (int shift = 24; shift >= 0; shift -= 8) {
    if (tid < 256) hist[tid] = 0u;
    __syncthreads();
    const unsigned himask = (shift == 24) ? 0u : (0xFFFFFFFFu << (shift + 8));
    const unsigned pref = sh_prefix;
    for (int i = tid; i < NN; i += 1024) {
      unsigned u = fkey(sc[i]);
      if ((u & himask) == (pref & himask))
        atomicAdd(&hist[(u >> shift) & 255u], 1u);
    }
    __syncthreads();
    if (tid == 0) {
      unsigned cum = 0;
      int d = 255;
      for (; d > 0; --d) {
        if (cum + hist[d] >= (unsigned)sh_rem) break;
        cum += hist[d];
      }
      sh_rem -= (int)cum;
      sh_prefix |= ((unsigned)d) << shift;
    }
    __syncthreads();
  }

  const unsigned T = sh_prefix;
  const int need = sh_rem;
  if (tid == 0) { s_base = 0; s_tie = 0; }
  __syncthreads();

  for (int chunk = 0; chunk < NN / 1024; ++chunk) {
    const int i = chunk * 1024 + tid;
    const unsigned u = fkey(sc[i]);
    const int e = (u == T) ? 1 : 0;
    const int g = (u > T) ? 1 : 0;

    sbuf[tid] = e; __syncthreads();
    for (int off = 1; off < 1024; off <<= 1) {
      int t = (tid >= off) ? sbuf[tid - off] : 0;
      __syncthreads();
      sbuf[tid] += t;
      __syncthreads();
    }
    const int einc = sbuf[tid];
    if (tid == 1023) s_tot_e = sbuf[1023];
    __syncthreads();

    const int keep = (g || (e && (s_tie + einc - 1 < need))) ? 1 : 0;

    sbuf[tid] = keep; __syncthreads();
    for (int off = 1; off < 1024; off <<= 1) {
      int t = (tid >= off) ? sbuf[tid - off] : 0;
      __syncthreads();
      sbuf[tid] += t;
      __syncthreads();
    }
    const int kinc = sbuf[tid];
    if (tid == 1023) s_tot_k = sbuf[1023];
    __syncthreads();

    if (keep) {
      int pos = s_base + kinc - 1;
      if (pos < KK) idx[b * KK + pos] = i;
    }
    __syncthreads();
    if (tid == 0) { s_base += s_tot_k; s_tie += s_tot_e; }
    __syncthreads();
  }
}

// ---------------------------------------------------------------------------
// Kernel 3: gather x_kept[row,:] = x[b, idx, :] * score  (float4 streaming)
// ---------------------------------------------------------------------------
__global__ __launch_bounds__(256) void gather_kernel(
    const float* __restrict__ x, const float* __restrict__ scores,
    const int* __restrict__ idx, float* __restrict__ out) {
  const int row = blockIdx.x;        // 0 .. B*KK-1
  const int b = row / KK;
  const int n = idx[row];
  const float s = scores[b * NN + n];
  const float4* xr = (const float4*)(x + ((size_t)b * NN + n) * DD);
  float4* o = (float4*)(out + (size_t)row * DD);
  float4 v = xr[threadIdx.x];
  v.x *= s; v.y *= s; v.z *= s; v.w *= s;
  o[threadIdx.x] = v;
}

// ---------------------------------------------------------------------------
// Kernels 4/5: sparsity_loss = -mean(|score - 0.5|), deterministic 2-stage.
// ---------------------------------------------------------------------------
__global__ __launch_bounds__(256) void loss_partial_kernel(
    const float* __restrict__ scores, float* __restrict__ partials) {
  __shared__ float red[256];
  const int tid = threadIdx.x;
  float s = 0.0f;
  for (int i = blockIdx.x * 256 + tid; i < ROWS; i += 64 * 256)
    s += fabsf(scores[i] - 0.5f);
  red[tid] = s; __syncthreads();
  for (int off = 128; off > 0; off >>= 1) {
    if (tid < off) red[tid] += red[tid + off];
    __syncthreads();
  }
  if (tid == 0) partials[blockIdx.x] = red[0];
}

__global__ __launch_bounds__(64) void loss_final_kernel(
    const float* __restrict__ partials, float* __restrict__ out_loss) {
  __shared__ float red[64];
  const int tid = threadIdx.x;
  red[tid] = partials[tid]; __syncthreads();
  for (int off = 32; off > 0; off >>= 1) {
    if (tid < off) red[tid] += red[tid + off];
    __syncthreads();
  }
  if (tid == 0) out_loss[0] = -red[0] / (float)ROWS;
}

// ---------------------------------------------------------------------------
extern "C" void kernel_launch(void* const* d_in, const int* in_sizes, int n_in,
                              void* d_out, int out_size, void* d_ws, size_t ws_size,
                              hipStream_t stream) {
  (void)in_sizes; (void)n_in; (void)out_size; (void)ws_size;
  const float* x  = (const float*)d_in[0];
  const float* W1 = (const float*)d_in[1];
  const float* b1 = (const float*)d_in[2];
  const float* W2 = (const float*)d_in[3];
  const float* b2 = (const float*)d_in[4];
  float* out = (float*)d_out;

  char* ws = (char*)d_ws;
  float*  scores   = (float*)ws;                          // 256 KB
  int*    idx      = (int*)(ws + 262144);                 // 184 KB
  float*  partials = (float*)(ws + 262144 + 183488);      // 256 B
  __bf16* W1bf     = (__bf16*)(ws + 458752);              // 512 KB (448K offset)

  convert_w1_kernel<<<1024, 256, 0, stream>>>(W1, W1bf);
  scores_kernel<<<ROWS / 128, 256, 0, stream>>>(x, W1bf, b1, W2, b2, scores);
  topk_kernel<<<BB, 1024, 0, stream>>>(scores, idx);
  gather_kernel<<<BB * KK, 256, 0, stream>>>(x, scores, idx, out);
  loss_partial_kernel<<<64, 256, 0, stream>>>(scores, partials);
  loss_final_kernel<<<1, 64, 0, stream>>>(partials, out + (size_t)BB * KK * DD);
}
